// Efficia1Block_45698452030315
// MI455X (gfx1250) — matrix operand order repaired and checked by reading, not compiled, not run on hardware
//
#include <hip/hip_runtime.h>
#include <hip/hip_bf16.h>
#include <math.h>
#include <stdint.h>

// ---------------------------------------------------------------------------
// CDNA5 (gfx1250) wave32 WMMA bf16 implementation of the Efficia1 block.
// Heavy GEMMs: double-buffered LDS tiles filled with GLOBAL_LOAD_ASYNC_TO_LDS
// (ASYNCcnt), computed with v_wmma_f32_16x16x32_bf16 (f32 accumulate).
// ---------------------------------------------------------------------------

typedef __bf16 bfx16 __attribute__((ext_vector_type(16)));
typedef __bf16 bfx8  __attribute__((ext_vector_type(8)));
typedef float  fx8   __attribute__((ext_vector_type(8)));

static __device__ inline fx8 wmma_bf16(bfx16 a, bfx16 b, fx8 c) {
  return __builtin_amdgcn_wmma_f32_16x16x32_bf16(
      /*neg_a=*/false, a, /*neg_b=*/false, b,
      /*c_mod=*/(short)0, c, /*reuse_a=*/false, /*reuse_b=*/false);
}

// Async DMA: 16 bytes per lane, global -> LDS (tracked by ASYNCcnt).
static __device__ inline void async_copy_b128(unsigned lds_off, const __bf16* g) {
  asm volatile("global_load_async_to_lds_b128 %0, %1, off"
               :: "v"(lds_off), "v"(g)
               : "memory");
}
static __device__ inline void wait_asynccnt0() {
  asm volatile("s_wait_asynccnt 0x0" ::: "memory");
}

// A-fragment (16x32 bf16, M x K). ISA layout: lane&15 = M row; lanes 0-15 hold
// K 0-7 / 16-23, lanes 16-31 hold K 8-15 / 24-31. p0 points at (m=0, k=0).
static __device__ inline bfx16 load_a_bf16(const __bf16* p0, int ld, int kBase) {
  const int lane = threadIdx.x & 31;
  const __bf16* p = p0 + (size_t)(lane & 15) * ld + kBase + ((lane >> 4) << 3);
  bfx8 lo = *(const bfx8*)p;
  bfx8 hi = *(const bfx8*)(p + 16);
  bfx16 r;
#pragma unroll
  for (int i = 0; i < 8; ++i) { r[i] = lo[i]; r[i + 8] = hi[i]; }
  return r;
}

// Same A-fragment but sourced from fp32 global memory with on-the-fly bf16 cast.
static __device__ inline bfx16 load_a_f32(const float* p0, int ld, int kBase) {
  const int lane = threadIdx.x & 31;
  const float* p = p0 + (size_t)(lane & 15) * ld + kBase + ((lane >> 4) << 3);
  bfx16 r;
#pragma unroll
  for (int i = 0; i < 8; ++i) { r[i] = (__bf16)p[i]; r[i + 8] = (__bf16)p[i + 16]; }
  return r;
}

// B-fragment (32x16 bf16, K x N) from an N-major [n][k] buffer (ld = k stride).
// ISA layout: lane&15 = N col; lanes 0-15 hold K 0-7 / 16-23, lanes 16-31 K 8-15 / 24-31.
static __device__ inline bfx16 load_b_nk(const __bf16* p0, int ld, int nBase, int kBase) {
  const int lane = threadIdx.x & 31;
  const __bf16* p = p0 + (size_t)(nBase + (lane & 15)) * ld + kBase + ((lane >> 4) << 3);
  bfx8 lo = *(const bfx8*)p;
  bfx8 hi = *(const bfx8*)(p + 16);
  bfx16 r;
#pragma unroll
  for (int i = 0; i < 8; ++i) { r[i] = lo[i]; r[i + 8] = hi[i]; }
  return r;
}

// ---------------------------------------------------------------------------
// Tiled GEMM: C[M,N] = A[M,K] @ B[K,N] (+ optional f32 addend).
// A is [M,K] bf16 row-major; Bt is the WEIGHT TRANSPOSED to [N,K] bf16 so both
// tiles are contiguous-row async copies and land in LDS already fragment-ready.
// 128x128 tile, 8 waves (2x4), K-step 64, double-buffered LDS, async pipeline.
// ---------------------------------------------------------------------------
template <typename OutT>
__global__ __launch_bounds__(256) void k_gemm(
    const __bf16* __restrict__ A, long lda,
    const __bf16* __restrict__ Bt, long ldb,   // [N,K]
    OutT* __restrict__ C, long ldc,
    const float* __restrict__ Add, long ldadd,
    int M, int N, int K, float alpha) {
  __shared__ __bf16 shA[2][128 * 64];   // [m][k] per stage, 16KB each
  __shared__ __bf16 shB[2][128 * 64];   // [n][k] per stage, 16KB each
  const int tid = threadIdx.x;
  const int wave = tid >> 5;
  const int wm = (wave >> 2) * 64;
  const int wn = (wave & 3) * 32;
  const int m0 = blockIdx.y * 128;
  const int n0 = blockIdx.x * 128;

  fx8 acc[4][2];
#pragma unroll
  for (int i = 0; i < 4; ++i)
#pragma unroll
    for (int j = 0; j < 2; ++j)
#pragma unroll
      for (int e = 0; e < 8; ++e) acc[i][j][e] = 0.f;

  auto compute_stage = [&](int buf) {
    const __bf16* sA = shA[buf];
    const __bf16* sB = shB[buf];
#pragma unroll
    for (int sub = 0; sub < 2; ++sub) {
      // Preload all fragments of this sub-step, then issue 8 back-to-back WMMAs
      bfx16 b0 = load_b_nk(sB, 64, wn, sub * 32);
      bfx16 b1 = load_b_nk(sB, 64, wn + 16, sub * 32);
      bfx16 a0 = load_a_bf16(sA + (size_t)(wm + 0) * 64, 64, sub * 32);
      bfx16 a1 = load_a_bf16(sA + (size_t)(wm + 16) * 64, 64, sub * 32);
      bfx16 a2 = load_a_bf16(sA + (size_t)(wm + 32) * 64, 64, sub * 32);
      bfx16 a3 = load_a_bf16(sA + (size_t)(wm + 48) * 64, 64, sub * 32);
      acc[0][0] = wmma_bf16(a0, b0, acc[0][0]);
      acc[0][1] = wmma_bf16(a0, b1, acc[0][1]);
      acc[1][0] = wmma_bf16(a1, b0, acc[1][0]);
      acc[1][1] = wmma_bf16(a1, b1, acc[1][1]);
      acc[2][0] = wmma_bf16(a2, b0, acc[2][0]);
      acc[2][1] = wmma_bf16(a2, b1, acc[2][1]);
      acc[3][0] = wmma_bf16(a3, b0, acc[3][0]);
      acc[3][1] = wmma_bf16(a3, b1, acc[3][1]);
    }
  };

  const bool aligned = (M % 128 == 0) && (N % 128 == 0) && (K % 64 == 0) &&
                       ((lda & 7) == 0) && ((ldb & 7) == 0);
  if (aligned) {
    // Async double-buffered pipeline. Each stage = 128x64 A + 128x64 B =
    // 2048 x 16B segments, 8 segments per thread, ASYNCcnt-tracked.
    const unsigned offA = (unsigned)(uintptr_t)(&shA[0][0]);
    const unsigned offB = (unsigned)(uintptr_t)(&shB[0][0]);
    auto issue_stage = [&](int t, int buf) {
      const unsigned bA = offA + (unsigned)buf * (128 * 64 * 2);
      const unsigned bB = offB + (unsigned)buf * (128 * 64 * 2);
      const int k0 = t * 64;
#pragma unroll
      for (int i = 0; i < 4; ++i) {
        int s = tid + i * 256;          // 0..1023
        int row = s >> 3, seg = s & 7;  // 8 x 16B segments per 64-elem row
        async_copy_b128(bA + (unsigned)(row * 128 + seg * 16),
                        A + (size_t)(m0 + row) * lda + k0 + seg * 8);
        async_copy_b128(bB + (unsigned)(row * 128 + seg * 16),
                        Bt + (size_t)(n0 + row) * ldb + k0 + seg * 8);
      }
    };
    const int T = K / 64;
    issue_stage(0, 0);
    for (int t = 0; t < T; ++t) {
      wait_asynccnt0();                      // own stage-t DMAs retired
      __syncthreads();                       // stage t visible; buf (t+1)&1 free
      if (t + 1 < T) issue_stage(t + 1, (t + 1) & 1);  // overlaps compute(t)
      compute_stage(t & 1);
    }
  } else {
    // Guarded synchronous fallback (edge shapes).
    for (int k0 = 0; k0 < K; k0 += 64) {
      __syncthreads();
#pragma unroll
      for (int it = 0; it < 32; ++it) {
        int idx = tid + it * 256;
        int r = idx >> 6, k = idx & 63;
        int gm = m0 + r, gk = k0 + k;
        shA[0][idx] = (gm < M && gk < K) ? A[(size_t)gm * lda + gk] : (__bf16)0.f;
        int gn = n0 + r;
        shB[0][idx] = (gn < N && gk < K) ? Bt[(size_t)gn * ldb + gk] : (__bf16)0.f;
      }
      __syncthreads();
      compute_stage(0);
    }
  }

  const int lane = tid & 31;
  const int nl = lane & 15;
  const int mh = (lane >> 4) << 3;
#pragma unroll
  for (int mt = 0; mt < 4; ++mt)
#pragma unroll
    for (int nt = 0; nt < 2; ++nt)
#pragma unroll
      for (int r = 0; r < 8; ++r) {
        int gm = m0 + wm + mt * 16 + r + mh;
        int gn = n0 + wn + nt * 16 + nl;
        if (gm < M && gn < N) {
          float v = acc[mt][nt][r] * alpha;
          if (Add) v += Add[(size_t)gm * ldadd + gn];
          C[(size_t)gm * ldc + gn] = (OutT)v;
        }
      }
}

// ---------------------------------------------------------------------------
// Local windowed attention: one block per (window, head, batch).
// Q,K,V pulled from the packed qkv fp32 buffer (cols = part*D + h*128 + d).
// ---------------------------------------------------------------------------
__global__ __launch_bounds__(256) void k_local_attn(const float* __restrict__ qkv,
                                                    float* __restrict__ O) {
  constexpr int DHc = 128, WINc = 128, Dc = 2048, Sc = 2048;
  __shared__ __bf16 shK[WINc * DHc];   // [key][d]  32KB
  __shared__ __bf16 shVt[DHc * WINc];  // [d][key]  32KB
  const int w = blockIdx.x, h = blockIdx.y, b = blockIdx.z;
  const int tid = threadIdx.x, wave = tid >> 5;
  const size_t tok0 = (size_t)b * Sc + (size_t)w * WINc;
  const float* qbase = qkv + tok0 * (3 * Dc) + h * DHc;
  const float* kbase = qbase + Dc;
  const float* vbase = qbase + 2 * Dc;
#pragma unroll
  for (int it = 0; it < 64; ++it) {
    int idx = tid + it * 256;
    int r = idx >> 7, d = idx & 127;
    shK[r * DHc + d] = (__bf16)kbase[(size_t)r * (3 * Dc) + d];
    shVt[d * WINc + r] = (__bf16)vbase[(size_t)r * (3 * Dc) + d];
  }
  __syncthreads();

  const int q0 = wave * 16;
  bfx16 aq[4];
#pragma unroll
  for (int kb = 0; kb < 4; ++kb)
    aq[kb] = load_a_f32(qbase + (size_t)q0 * (3 * Dc), 3 * Dc, kb * 32);

  fx8 sc[8];
#pragma unroll
  for (int nt = 0; nt < 8; ++nt)
#pragma unroll
    for (int e = 0; e < 8; ++e) sc[nt][e] = 0.f;
#pragma unroll
  for (int kb = 0; kb < 4; ++kb)
#pragma unroll
    for (int nt = 0; nt < 8; ++nt)
      sc[nt] = wmma_bf16(aq[kb], load_b_nk(shK, DHc, nt * 16, kb * 32), sc[nt]);

  const float scale = 0.08838834764831845f;  // 1/sqrt(128)
  float inv[8];
#pragma unroll
  for (int r = 0; r < 8; ++r) {
    float m = -1e30f;
#pragma unroll
    for (int nt = 0; nt < 8; ++nt) { sc[nt][r] *= scale; m = fmaxf(m, sc[nt][r]); }
#pragma unroll
    for (int off = 8; off; off >>= 1) m = fmaxf(m, __shfl_xor(m, off));
    float s = 0.f;
#pragma unroll
    for (int nt = 0; nt < 8; ++nt) { float p = __expf(sc[nt][r] - m); sc[nt][r] = p; s += p; }
#pragma unroll
    for (int off = 8; off; off >>= 1) s += __shfl_xor(s, off);
    inv[r] = 1.f / s;
  }

  __syncthreads();                 // done reading shK -> reuse as P
  __bf16* shP = shK;               // [q][key]
  const int lane = tid & 31, nl = lane & 15, mh = (lane >> 4) << 3;
#pragma unroll
  for (int nt = 0; nt < 8; ++nt)
#pragma unroll
    for (int r = 0; r < 8; ++r)
      shP[(size_t)(q0 + r + mh) * WINc + nt * 16 + nl] = (__bf16)(sc[nt][r] * inv[r]);
  __syncthreads();

  fx8 oa[8];
#pragma unroll
  for (int nt = 0; nt < 8; ++nt)
#pragma unroll
    for (int e = 0; e < 8; ++e) oa[nt][e] = 0.f;
#pragma unroll
  for (int kb = 0; kb < 4; ++kb) {
    bfx16 ap = load_a_bf16(shP + (size_t)q0 * WINc, WINc, kb * 32);
#pragma unroll
    for (int nt = 0; nt < 8; ++nt)
      oa[nt] = wmma_bf16(ap, load_b_nk(shVt, WINc, nt * 16, kb * 32), oa[nt]);
  }
  float* obase = O + (tok0 + q0) * Dc + h * DHc;
#pragma unroll
  for (int nt = 0; nt < 8; ++nt)
#pragma unroll
    for (int r = 0; r < 8; ++r)
      obase[(size_t)(r + mh) * Dc + nt * 16 + nl] = oa[nt][r];
}

// ---------------------------------------------------------------------------
// Global-memory read attention: S queries over M=64 keys per (b,h).
// ---------------------------------------------------------------------------
__global__ __launch_bounds__(256) void k_global_attn(const __bf16* __restrict__ qg,
                                                     const __bf16* __restrict__ kg,
                                                     const __bf16* __restrict__ vg,
                                                     float* __restrict__ R) {
  constexpr int Mk = 64, DHc = 128, Dc = 2048, Sc = 2048;
  __shared__ __bf16 shK[Mk * DHc];
  __shared__ __bf16 shVt[DHc * Mk];
  __shared__ __bf16 shP[128 * Mk];
  const int qb = blockIdx.x, h = blockIdx.y, b = blockIdx.z;
  const int tid = threadIdx.x, wave = tid >> 5;
#pragma unroll
  for (int it = 0; it < 32; ++it) {
    int idx = tid + it * 256;
    int r = idx >> 7, d = idx & 127;
    size_t g = ((size_t)b * Mk + r) * Dc + h * DHc + d;
    shK[r * DHc + d] = kg[g];
    shVt[d * Mk + r] = vg[g];
  }
  __syncthreads();

  const int q0 = wave * 16;
  const size_t qrow0 = (size_t)b * Sc + (size_t)qb * 128 + q0;
  bfx16 aq[4];
#pragma unroll
  for (int kb = 0; kb < 4; ++kb)
    aq[kb] = load_a_bf16(qg + qrow0 * Dc + h * DHc, Dc, kb * 32);

  fx8 sc[4];
#pragma unroll
  for (int nt = 0; nt < 4; ++nt)
#pragma unroll
    for (int e = 0; e < 8; ++e) sc[nt][e] = 0.f;
#pragma unroll
  for (int kb = 0; kb < 4; ++kb)
#pragma unroll
    for (int nt = 0; nt < 4; ++nt)
      sc[nt] = wmma_bf16(aq[kb], load_b_nk(shK, DHc, nt * 16, kb * 32), sc[nt]);

  const float scale = 0.08838834764831845f;
  float inv[8];
#pragma unroll
  for (int r = 0; r < 8; ++r) {
    float m = -1e30f;
#pragma unroll
    for (int nt = 0; nt < 4; ++nt) { sc[nt][r] *= scale; m = fmaxf(m, sc[nt][r]); }
#pragma unroll
    for (int off = 8; off; off >>= 1) m = fmaxf(m, __shfl_xor(m, off));
    float s = 0.f;
#pragma unroll
    for (int nt = 0; nt < 4; ++nt) { float p = __expf(sc[nt][r] - m); sc[nt][r] = p; s += p; }
#pragma unroll
    for (int off = 8; off; off >>= 1) s += __shfl_xor(s, off);
    inv[r] = 1.f / s;
  }
  const int lane = tid & 31, nl = lane & 15, mh = (lane >> 4) << 3;
#pragma unroll
  for (int nt = 0; nt < 4; ++nt)
#pragma unroll
    for (int r = 0; r < 8; ++r)
      shP[(size_t)(q0 + r + mh) * Mk + nt * 16 + nl] = (__bf16)(sc[nt][r] * inv[r]);
  __syncthreads();

  fx8 oa[8];
#pragma unroll
  for (int nt = 0; nt < 8; ++nt)
#pragma unroll
    for (int e = 0; e < 8; ++e) oa[nt][e] = 0.f;
#pragma unroll
  for (int kb = 0; kb < 2; ++kb) {
    bfx16 ap = load_a_bf16(shP + (size_t)q0 * Mk, Mk, kb * 32);
#pragma unroll
    for (int nt = 0; nt < 8; ++nt)
      oa[nt] = wmma_bf16(ap, load_b_nk(shVt, Mk, nt * 16, kb * 32), oa[nt]);
  }
  float* obase = R + qrow0 * Dc + h * DHc;
#pragma unroll
  for (int nt = 0; nt < 8; ++nt)
#pragma unroll
    for (int r = 0; r < 8; ++r)
      obase[(size_t)(r + mh) * Dc + nt * 16 + nl] = oa[nt][r];
}

// ---------------------------------------------------------------------------
// Memory-write attention (flash style): 64 queries over S=2048 keys per (b,h),
// online softmax over 64-key chunks. 4 waves, 16 query rows each.
// ---------------------------------------------------------------------------
__global__ __launch_bounds__(128) void k_mem_attn(const __bf16* __restrict__ qm,
                                                  const __bf16* __restrict__ km,
                                                  const __bf16* __restrict__ vm,
                                                  float* __restrict__ WR) {
  constexpr int Mq = 64, CH = 64, DHc = 128, Dc = 2048, Sc = 2048;
  __shared__ __bf16 shK[CH * DHc];
  __shared__ __bf16 shVt[DHc * CH];
  __shared__ __bf16 shP[Mq * CH];
  const int h = blockIdx.x, b = blockIdx.y;
  const int tid = threadIdx.x, wave = tid >> 5;
  const int q0 = wave * 16;
  const int lane = tid & 31, nl = lane & 15, mh = (lane >> 4) << 3;
  const float scale = 0.08838834764831845f;

  bfx16 aq[4];
#pragma unroll
  for (int kb = 0; kb < 4; ++kb)
    aq[kb] = load_a_bf16(qm + ((size_t)b * Mq + q0) * Dc + h * DHc, Dc, kb * 32);

  fx8 oa[8];
#pragma unroll
  for (int nt = 0; nt < 8; ++nt)
#pragma unroll
    for (int e = 0; e < 8; ++e) oa[nt][e] = 0.f;
  float runM[8], runS[8];
#pragma unroll
  for (int r = 0; r < 8; ++r) { runM[r] = -1e30f; runS[r] = 0.f; }

  for (int c = 0; c < Sc / CH; ++c) {
    __syncthreads();
#pragma unroll
    for (int it = 0; it < 64; ++it) {
      int idx = tid + it * 128;
      int r = idx >> 7, d = idx & 127;
      size_t row = ((size_t)b * Sc + (size_t)c * CH + r) * Dc + h * DHc + d;
      shK[r * DHc + d] = km[row];
      shVt[d * CH + r] = vm[row];
    }
    __syncthreads();

    fx8 sc4[4];
#pragma unroll
    for (int nt = 0; nt < 4; ++nt)
#pragma unroll
      for (int e = 0; e < 8; ++e) sc4[nt][e] = 0.f;
#pragma unroll
    for (int kb = 0; kb < 4; ++kb)
#pragma unroll
      for (int nt = 0; nt < 4; ++nt)
        sc4[nt] = wmma_bf16(aq[kb], load_b_nk(shK, DHc, nt * 16, kb * 32), sc4[nt]);

#pragma unroll
    for (int r = 0; r < 8; ++r) {
      float m = -1e30f;
#pragma unroll
      for (int nt = 0; nt < 4; ++nt) { sc4[nt][r] *= scale; m = fmaxf(m, sc4[nt][r]); }
#pragma unroll
      for (int off = 8; off; off >>= 1) m = fmaxf(m, __shfl_xor(m, off));
      float nm = fmaxf(runM[r], m);
      float corr = __expf(runM[r] - nm);
      float s = 0.f;
#pragma unroll
      for (int nt = 0; nt < 4; ++nt) { float p = __expf(sc4[nt][r] - nm); sc4[nt][r] = p; s += p; }
#pragma unroll
      for (int off = 8; off; off >>= 1) s += __shfl_xor(s, off);
      runS[r] = runS[r] * corr + s;
      runM[r] = nm;
#pragma unroll
      for (int nt = 0; nt < 8; ++nt) oa[nt][r] *= corr;
    }
#pragma unroll
    for (int nt = 0; nt < 4; ++nt)
#pragma unroll
      for (int r = 0; r < 8; ++r)
        shP[(size_t)(q0 + r + mh) * CH + nt * 16 + nl] = (__bf16)sc4[nt][r];
    __syncthreads();
#pragma unroll
    for (int kb = 0; kb < 2; ++kb) {
      bfx16 ap = load_a_bf16(shP + (size_t)q0 * CH, CH, kb * 32);
#pragma unroll
      for (int nt = 0; nt < 8; ++nt)
        oa[nt] = wmma_bf16(ap, load_b_nk(shVt, CH, nt * 16, kb * 32), oa[nt]);
    }
  }
  float* wbase = WR + ((size_t)b * Mq + q0) * Dc + h * DHc;
#pragma unroll
  for (int nt = 0; nt < 8; ++nt)
#pragma unroll
    for (int r = 0; r < 8; ++r)
      wbase[(size_t)(r + mh) * Dc + nt * 16 + nl] = oa[nt][r] / runS[r];
}

// ---------------------------------------------------------------------------
// Elementwise / small helper kernels.
// ---------------------------------------------------------------------------
__global__ void k_cast_bf16(const float* __restrict__ x, __bf16* __restrict__ y, size_t n) {
  size_t i = (size_t)blockIdx.x * blockDim.x + threadIdx.x;
  if (i < n) y[i] = (__bf16)x[i];
}

// Transpose-cast f32 [K,N] -> bf16 [N,K], 32x32 LDS tile for coalescing.
__global__ __launch_bounds__(256) void k_cast_t_bf16(const float* __restrict__ src,
                                                     __bf16* __restrict__ dst,
                                                     int K, int N) {
  __shared__ float t[32][33];
  const int kb = blockIdx.x * 32, nb = blockIdx.y * 32;
  const int tx = threadIdx.x & 31, ty = threadIdx.x >> 5;  // ty 0..7
#pragma unroll
  for (int i = ty; i < 32; i += 8) {
    int k = kb + i, n = nb + tx;
    t[i][tx] = (k < K && n < N) ? src[(size_t)k * N + n] : 0.f;
  }
  __syncthreads();
#pragma unroll
  for (int i = ty; i < 32; i += 8) {
    int n = nb + i, k = kb + tx;
    if (n < N && k < K) dst[(size_t)n * K + k] = (__bf16)t[tx][i];
  }
}

__global__ __launch_bounds__(256) void k_rmsnorm(const float* __restrict__ X,
                                                 const float* __restrict__ g,
                                                 float* __restrict__ outF,
                                                 __bf16* __restrict__ outB,
                                                 int D, int addResid) {
  __shared__ float red[8];
  const float* x = X + (size_t)blockIdx.x * D;
  float ss = 0.f;
  for (int i = threadIdx.x; i < D; i += 256) { float v = x[i]; ss += v * v; }
#pragma unroll
  for (int off = 16; off; off >>= 1) ss += __shfl_xor(ss, off);
  if ((threadIdx.x & 31) == 0) red[threadIdx.x >> 5] = ss;
  __syncthreads();
  if (threadIdx.x == 0) {
    float t = 0.f;
#pragma unroll
    for (int i = 0; i < 8; ++i) t += red[i];
    red[0] = rsqrtf(t / (float)D + 1e-6f);
  }
  __syncthreads();
  float scl = red[0];
  for (int i = threadIdx.x; i < D; i += 256) {
    float v = x[i];
    float y = v * scl * g[i];
    float o = addResid ? v + y : y;
    if (outF) outF[(size_t)blockIdx.x * D + i] = o;
    if (outB) outB[(size_t)blockIdx.x * D + i] = (__bf16)o;
  }
}

__global__ void k_gate_combine(const float* __restrict__ h2, const float* __restrict__ gt,
                               const float* __restrict__ go, float* __restrict__ out, size_t n) {
  size_t i = (size_t)blockIdx.x * blockDim.x + threadIdx.x;
  if (i < n) out[i] = h2[i] + (1.f / (1.f + __expf(-gt[i]))) * go[i];
}

__global__ void k_gelu_cast(const float* __restrict__ x, __bf16* __restrict__ y, size_t n) {
  size_t i = (size_t)blockIdx.x * blockDim.x + threadIdx.x;
  if (i < n) {
    float v = x[i];
    float c = 0.7978845608028654f * (v + 0.044715f * v * v * v);
    y[i] = (__bf16)(0.5f * v * (1.f + tanhf(c)));
  }
}

__global__ void k_meanpool(const float* __restrict__ x, float* __restrict__ pooled) {
  size_t i = (size_t)blockIdx.x * blockDim.x + threadIdx.x;
  if (i >= 4ull * 2048ull) return;
  int b = (int)(i / 2048), d = (int)(i % 2048);
  float s = 0.f;
  for (int t = 0; t < 2048; ++t) s += x[((size_t)b * 2048 + t) * 2048 + d];
  pooled[i] = s * (1.f / 2048.f);
}

__global__ void k_state(const float* __restrict__ pooled, const float* __restrict__ wcz,
                        const float* __restrict__ wcg, const float* __restrict__ cs,
                        float* __restrict__ ns) {
  size_t i = (size_t)blockIdx.x * blockDim.x + threadIdx.x;
  if (i >= 4ull * 256ull) return;
  int b = (int)(i >> 8), c = (int)(i & 255);
  const float* p = pooled + (size_t)b * 2048;
  float az = 0.f, ag = 0.f;
  for (int k = 0; k < 2048; ++k) {
    float pv = p[k];
    az += pv * wcz[(size_t)k * 256 + c];
    ag += pv * wcg[(size_t)k * 256 + c];
  }
  float z = tanhf(az);
  float gcv = 1.f / (1.f + __expf(-ag));
  ns[i] = gcv * cs[i] + (1.f - gcv) * z;
}

__global__ void k_small_mm(const float* __restrict__ A, const float* __restrict__ W,
                           float* __restrict__ C, int Bn, int K, int N, long ldw) {
  size_t i = (size_t)blockIdx.x * blockDim.x + threadIdx.x;
  if (i >= (size_t)Bn * N) return;
  int b = (int)(i / N), n = (int)(i % N);
  float s = 0.f;
  for (int k = 0; k < K; ++k) s += A[(size_t)b * K + k] * W[(size_t)k * ldw + n];
  C[i] = s;
}

__global__ void k_bcast_add(float* __restrict__ fused, const float* __restrict__ spf) {
  size_t i = (size_t)blockIdx.x * blockDim.x + threadIdx.x;
  if (i >= 4ull * 2048ull * 2048ull) return;
  int b = (int)(i / (2048ull * 2048ull));
  int d = (int)(i % 2048ull);
  fused[i] += spf[(size_t)b * 2048 + d];
}

// ---------------------------------------------------------------------------
// Host orchestration
// ---------------------------------------------------------------------------
extern "C" void kernel_launch(void* const* d_in, const int* in_sizes, int n_in,
                              void* d_out, int out_size, void* d_ws, size_t ws_size,
                              hipStream_t stream) {
  (void)in_sizes; (void)n_in; (void)out_size; (void)ws_size;
  const float* x     = (const float*)d_in[0];
  const float* gm    = (const float*)d_in[1];
  const float* cs    = (const float*)d_in[2];
  const float* g1    = (const float*)d_in[3];
  const float* g2    = (const float*)d_in[4];
  const float* g3    = (const float*)d_in[5];
  const float* g4    = (const float*)d_in[6];
  const float* w_qkv = (const float*)d_in[7];
  const float* w_lo  = (const float*)d_in[8];
  const float* w_gq  = (const float*)d_in[9];
  const float* w_gk  = (const float*)d_in[10];
  const float* w_gv  = (const float*)d_in[11];
  const float* w_go  = (const float*)d_in[12];
  const float* w_gate= (const float*)d_in[13];
  const float* w_mq  = (const float*)d_in[14];
  const float* w_mk  = (const float*)d_in[15];
  const float* w_mv  = (const float*)d_in[16];
  const float* w_mo  = (const float*)d_in[17];
  const float* w_cz  = (const float*)d_in[18];
  const float* w_cg  = (const float*)d_in[19];
  const float* w_sp  = (const float*)d_in[20];
  const float* w_fuse= (const float*)d_in[21];
  const float* w_ff1 = (const float*)d_in[22];
  const float* w_ff2 = (const float*)d_in[23];

  const long Bc = 4, Sc = 2048, Dc = 2048, Mc = 64, FFc = 8192;
  const long BS = Bc * Sc;

  float* outO     = (float*)d_out;
  float* outMem   = outO + (size_t)BS * Dc;
  float* outState = outMem + (size_t)Bc * Mc * Dc;

  char* ws = (char*)d_ws;
  size_t cur = 0;
  auto alloc = [&](size_t bytes) -> void* {
    cur = (cur + 255) & ~(size_t)255;
    void* p = ws + cur;
    cur += bytes;
    return p;
  };

  // Transposed bf16 weights: [N,K] so GEMM B-tiles are contiguous async copies
  __bf16* wqkvT  = (__bf16*)alloc((size_t)3 * Dc * Dc * 2);   // [6144,2048]
  __bf16* wloT   = (__bf16*)alloc((size_t)Dc * Dc * 2);
  __bf16* wgqT   = (__bf16*)alloc((size_t)Dc * Dc * 2);
  __bf16* wgkT   = (__bf16*)alloc((size_t)Dc * Dc * 2);
  __bf16* wgvT   = (__bf16*)alloc((size_t)Dc * Dc * 2);
  __bf16* wgoT   = (__bf16*)alloc((size_t)Dc * Dc * 2);
  __bf16* wgateT = (__bf16*)alloc((size_t)Dc * Dc * 2);
  __bf16* wmqT   = (__bf16*)alloc((size_t)Dc * Dc * 2);
  __bf16* wmkT   = (__bf16*)alloc((size_t)Dc * Dc * 2);
  __bf16* wmvT   = (__bf16*)alloc((size_t)Dc * Dc * 2);
  __bf16* wmoT   = (__bf16*)alloc((size_t)Dc * Dc * 2);
  __bf16* wfuse0T= (__bf16*)alloc((size_t)Dc * Dc * 2);
  __bf16* wfuse1T= (__bf16*)alloc((size_t)Dc * Dc * 2);
  __bf16* wff1T  = (__bf16*)alloc((size_t)Dc * FFc * 2);      // [8192,2048]
  __bf16* wff2T  = (__bf16*)alloc((size_t)FFc * Dc * 2);      // [2048,8192]

  // activations
  __bf16* h1_b   = (__bf16*)alloc((size_t)BS * Dc * 2);
  float*  qkvF   = (float*)alloc((size_t)BS * 3 * Dc * 4);
  float*  oF     = (float*)alloc((size_t)BS * Dc * 4);
  __bf16* o_b    = (__bf16*)alloc((size_t)BS * Dc * 2);
  float*  lcp    = (float*)alloc((size_t)BS * Dc * 4);
  __bf16* lcp_b  = (__bf16*)alloc((size_t)BS * Dc * 2);
  float*  h2F    = (float*)alloc((size_t)BS * Dc * 4);
  __bf16* h2_b   = (__bf16*)alloc((size_t)BS * Dc * 2);
  __bf16* gm_b   = (__bf16*)alloc((size_t)Bc * Mc * Dc * 2);
  __bf16* qg_b   = (__bf16*)alloc((size_t)BS * Dc * 2);
  __bf16* kg_b   = (__bf16*)alloc((size_t)Bc * Mc * Dc * 2);
  __bf16* vg_b   = (__bf16*)alloc((size_t)Bc * Mc * Dc * 2);
  float*  readF  = (float*)alloc((size_t)BS * Dc * 4);
  __bf16* read_b = (__bf16*)alloc((size_t)BS * Dc * 2);
  float*  gateF  = (float*)alloc((size_t)BS * Dc * 4);
  float*  goF    = (float*)alloc((size_t)BS * Dc * 4);
  float*  gmgF   = (float*)alloc((size_t)BS * Dc * 4);
  __bf16* gmg_b  = (__bf16*)alloc((size_t)BS * Dc * 2);
  __bf16* qm_b   = (__bf16*)alloc((size_t)Bc * Mc * Dc * 2);
  __bf16* km_b   = (__bf16*)alloc((size_t)BS * Dc * 2);
  __bf16* vm_b   = (__bf16*)alloc((size_t)BS * Dc * 2);
  float*  wrF    = (float*)alloc((size_t)Bc * Mc * Dc * 4);
  __bf16* wr_b   = (__bf16*)alloc((size_t)Bc * Mc * Dc * 2);
  float*  pooled = (float*)alloc((size_t)Bc * Dc * 4);
  float*  spv    = (float*)alloc((size_t)Bc * Dc * 4);
  float*  spf    = (float*)alloc((size_t)Bc * Dc * 4);
  float*  fusedF = (float*)alloc((size_t)BS * Dc * 4);
  float*  hF     = (float*)alloc((size_t)BS * Dc * 4);
  __bf16* h4_b   = (__bf16*)alloc((size_t)BS * Dc * 2);
  float*  ffaF   = (float*)alloc((size_t)BS * FFc * 4);
  __bf16* ffg_b  = (__bf16*)alloc((size_t)BS * FFc * 2);

  auto cast = [&](const float* src, __bf16* dst, size_t n) {
    k_cast_bf16<<<(unsigned)((n + 255) / 256), 256, 0, stream>>>(src, dst, n);
  };
  auto castT = [&](const float* src, __bf16* dst, int K, int N) {
    dim3 grid((K + 31) / 32, (N + 31) / 32, 1);
    k_cast_t_bf16<<<grid, 256, 0, stream>>>(src, dst, K, N);
  };
  auto gemmF = [&](const __bf16* A, long lda, const __bf16* Bt, long ldb, float* C,
                   long ldc, const float* add, long ldadd, int M, int N, int K) {
    dim3 grid((N + 127) / 128, (M + 127) / 128, 1);
    k_gemm<float><<<grid, 256, 0, stream>>>(A, lda, Bt, ldb, C, ldc, add, ldadd,
                                            M, N, K, 1.f);
  };
  auto gemmB = [&](const __bf16* A, long lda, const __bf16* Bt, long ldb, __bf16* C,
                   long ldc, int M, int N, int K) {
    dim3 grid((N + 127) / 128, (M + 127) / 128, 1);
    k_gemm<__bf16><<<grid, 256, 0, stream>>>(A, lda, Bt, ldb, C, ldc,
                                             (const float*)nullptr, 0, M, N, K, 1.f);
  };

  // Weight transpose-casts (once per call; f32 [K,N] -> bf16 [N,K])
  castT(w_qkv,  wqkvT,  (int)Dc, (int)(3 * Dc));
  castT(w_lo,   wloT,   (int)Dc, (int)Dc);
  castT(w_gq,   wgqT,   (int)Dc, (int)Dc);
  castT(w_gk,   wgkT,   (int)Dc, (int)Dc);
  castT(w_gv,   wgvT,   (int)Dc, (int)Dc);
  castT(w_go,   wgoT,   (int)Dc, (int)Dc);
  castT(w_gate, wgateT, (int)Dc, (int)Dc);
  castT(w_mq,   wmqT,   (int)Dc, (int)Dc);
  castT(w_mk,   wmkT,   (int)Dc, (int)Dc);
  castT(w_mv,   wmvT,   (int)Dc, (int)Dc);
  castT(w_mo,   wmoT,   (int)Dc, (int)Dc);
  castT(w_fuse,                       wfuse0T, (int)Dc, (int)Dc);
  castT(w_fuse + (size_t)Dc * Dc,     wfuse1T, (int)Dc, (int)Dc);
  castT(w_ff1,  wff1T,  (int)Dc, (int)FFc);
  castT(w_ff2,  wff2T,  (int)FFc, (int)Dc);
  cast(gm, gm_b, (size_t)Bc * Mc * Dc);

  // 1) Local windowed attention branch
  k_rmsnorm<<<(unsigned)BS, 256, 0, stream>>>(x, g1, nullptr, h1_b, (int)Dc, 0);
  gemmF(h1_b, Dc, wqkvT, Dc, qkvF, 3 * Dc, nullptr, 0, (int)BS, (int)(3 * Dc), (int)Dc);
  k_local_attn<<<dim3(16, 16, 4), 256, 0, stream>>>(qkvF, oF);
  cast(oF, o_b, (size_t)BS * Dc);
  gemmF(o_b, Dc, wloT, Dc, lcp, Dc, x, Dc, (int)BS, (int)Dc, (int)Dc);
  k_rmsnorm<<<(unsigned)BS, 256, 0, stream>>>(lcp, g2, h2F, h2_b, (int)Dc, 0);
  cast(lcp, lcp_b, (size_t)BS * Dc);

  // 2) Global-memory read attention + gate
  gemmB(h2_b, Dc, wgqT, Dc, qg_b, Dc, (int)BS, (int)Dc, (int)Dc);
  gemmB(gm_b, Dc, wgkT, Dc, kg_b, Dc, (int)(Bc * Mc), (int)Dc, (int)Dc);
  gemmB(gm_b, Dc, wgvT, Dc, vg_b, Dc, (int)(Bc * Mc), (int)Dc, (int)Dc);
  k_global_attn<<<dim3(16, 16, 4), 256, 0, stream>>>(qg_b, kg_b, vg_b, readF);
  cast(readF, read_b, (size_t)BS * Dc);
  gemmF(h2_b, Dc, wgateT, Dc, gateF, Dc, nullptr, 0, (int)BS, (int)Dc, (int)Dc);
  gemmF(read_b, Dc, wgoT, Dc, goF, Dc, nullptr, 0, (int)BS, (int)Dc, (int)Dc);
  k_gate_combine<<<(unsigned)(((size_t)BS * Dc + 255) / 256), 256, 0, stream>>>(
      h2F, gateF, goF, gmgF, (size_t)BS * Dc);
  cast(gmgF, gmg_b, (size_t)BS * Dc);

  // 3) Memory-write attention -> new_memory
  gemmB(gm_b, Dc, wmqT, Dc, qm_b, Dc, (int)(Bc * Mc), (int)Dc, (int)Dc);
  gemmB(h2_b, Dc, wmkT, Dc, km_b, Dc, (int)BS, (int)Dc, (int)Dc);
  gemmB(h2_b, Dc, wmvT, Dc, vm_b, Dc, (int)BS, (int)Dc, (int)Dc);
  k_mem_attn<<<dim3(16, 4), 128, 0, stream>>>(qm_b, km_b, vm_b, wrF);
  cast(wrF, wr_b, (size_t)Bc * Mc * Dc);
  gemmF(wr_b, Dc, wmoT, Dc, outMem, Dc, gm, Dc, (int)(Bc * Mc), (int)Dc, (int)Dc);

  // 4) Compressed state -> new_state, sp
  k_meanpool<<<(unsigned)((4 * 2048 + 255) / 256), 256, 0, stream>>>(gmgF, pooled);
  k_state<<<(unsigned)((1024 + 255) / 256), 256, 0, stream>>>(pooled, w_cz, w_cg, cs, outState);
  k_small_mm<<<(unsigned)((4 * 2048 + 255) / 256), 256, 0, stream>>>(
      outState, w_sp, spv, 4, 256, 2048, 2048);
  k_small_mm<<<(unsigned)((4 * 2048 + 255) / 256), 256, 0, stream>>>(
      spv, w_fuse + (size_t)2 * Dc * Dc, spf, 4, 2048, 2048, 2048);

  // 5) Fuse (concat GEMM as two accumulating GEMMs + broadcast add)
  gemmF(lcp_b, Dc, wfuse0T, Dc, fusedF, Dc, nullptr, 0, (int)BS, (int)Dc, (int)Dc);
  gemmF(gmg_b, Dc, wfuse1T, Dc, fusedF, Dc, fusedF, Dc, (int)BS, (int)Dc, (int)Dc);
  k_bcast_add<<<(unsigned)(((size_t)BS * Dc + 255) / 256), 256, 0, stream>>>(fusedF, spf);

  // 6) Output norm + FFN
  k_rmsnorm<<<(unsigned)BS, 256, 0, stream>>>(fusedF, g3, hF, nullptr, (int)Dc, 1);
  k_rmsnorm<<<(unsigned)BS, 256, 0, stream>>>(hF, g4, nullptr, h4_b, (int)Dc, 0);
  gemmF(h4_b, Dc, wff1T, Dc, ffaF, FFc, nullptr, 0, (int)BS, (int)FFc, (int)Dc);
  k_gelu_cast<<<(unsigned)(((size_t)BS * FFc + 255) / 256), 256, 0, stream>>>(
      ffaF, ffg_b, (size_t)BS * FFc);
  gemmF(ffg_b, FFc, wff2T, FFc, outO, Dc, hF, Dc, (int)BS, (int)Dc, (int)FFc);
}
